// AR_46660524704228
// MI455X (gfx1250) — compile-verified, matrix operand order checked by reading, not run on hardware
//
#include <hip/hip_runtime.h>

typedef __attribute__((ext_vector_type(16))) _Float16 v16h;
typedef __attribute__((ext_vector_type(8)))  _Float16 v8h;
typedef __attribute__((ext_vector_type(2)))  _Float16 v2h;
typedef __attribute__((ext_vector_type(8)))  float    v8f;

static constexpr int Bn = 32;     // batch
static constexpr int Sn = 8192;   // seq len
static constexpr int Dn = 32;     // hidden
static constexpr int Cn = 256;    // output categories
static constexpr int G4 = 4 * Dn; // 128 gate columns

__device__ __forceinline__ float fast_sigmoid(float x) {
  return __builtin_amdgcn_rcpf(1.0f + __expf(-x));
}
__device__ __forceinline__ float fast_tanh(float x) {
  // tanh(x) = 1 - 2/(exp(2x)+1)
  return 1.0f - 2.0f * __builtin_amdgcn_rcpf(1.0f + __expf(2.0f * x));
}
// LDS-only fence + split workgroup barrier (avoids __syncthreads()'s
// s_wait_storecnt: global hs stores must NOT gate the per-step barrier).
__device__ __forceinline__ void lds_fence_signal() {
  asm volatile("s_wait_dscnt 0\n\ts_barrier_signal -1" ::: "memory");
}
__device__ __forceinline__ void bar_wait() {
  asm volatile("s_barrier_wait -1" ::: "memory");
}

// ---------------------------------------------------------------------------
// Prep: rank-1 coefficients u,v,w0 and f16 WMMA-B-operand swizzles of Wh, W_out.
// 16-bit A/B operand layout (ISA 7.12.2): lane L, half h ->
//   rowcol = L & 15 ; K = 8*(L>>4) + 16*(h>>3) + (h&7)
// Wh B-tiles are COLUMN-PERMUTED: tile column n of wave w -> gate column
// (n&3)*32 + 4w + (n>>2), so wave w's C tile holds all four gates
// (i,f,g,o contiguous) for hidden units [4w,4w+4) -> wave-private gate update.
// ---------------------------------------------------------------------------
__global__ void __launch_bounds__(256) ar_prep(
    const float* __restrict__ bos, const float* __restrict__ W_in,
    const float* __restrict__ b_in, const float* __restrict__ Wx,
    const float* __restrict__ Wh, const float* __restrict__ b_lstm,
    const float* __restrict__ W_out,
    float* __restrict__ u, float* __restrict__ v, float* __restrict__ w0,
    _Float16* __restrict__ WhB, _Float16* __restrict__ WoutB) {
  const int tid = threadIdx.x;
  if (tid < G4) {
    float su = 0.f, sv = 0.f, s0 = 0.f;
    for (int d = 0; d < Dn; ++d) {
      const float wx = Wx[d * G4 + tid];
      su = fmaf(W_in[d], wx, su);
      sv = fmaf(b_in[d], wx, sv);
      s0 = fmaf(bos[d], wx, s0);
    }
    u[tid]  = su;
    v[tid]  = sv + b_lstm[tid];
    w0[tid] = s0 + b_lstm[tid];
  }
  for (int i = tid; i < 8 * 512; i += 256) {
    const int w = i >> 9, L = (i >> 4) & 31, h = i & 15;
    const int n = L & 15;
    const int k = ((L >> 4) * 8) + ((h >> 3) * 16) + (h & 7);
    const int kg = (n & 3) * Dn + 4 * w + (n >> 2);  // gate col permutation
    WhB[i] = (_Float16)Wh[k * G4 + kg];
  }
  for (int i = tid; i < 16 * 512; i += 256) {
    const int nt = i >> 9, L = (i >> 4) & 31, h = i & 15;
    const int n = L & 15;
    const int k = ((L >> 4) * 8) + ((h >> 3) * 16) + (h & 7);
    WoutB[i] = (_Float16)W_out[k * Cn + nt * 16 + n];
  }
}

// x (B,S) -> xT (S,B): per-step x values become one contiguous 32B row slice.
__global__ void __launch_bounds__(256) ar_xt(const float* __restrict__ x,
                                             float* __restrict__ xT) {
  const int idx = blockIdx.x * 256 + threadIdx.x;  // 0 .. B*S-1
  const int b = idx >> 13;                         // /Sn
  const int s = idx & (Sn - 1);
  xT[s * Bn + b] = x[idx];                         // coalesced reads
}

// ---------------------------------------------------------------------------
// Recurrence: 4 blocks x 8 batches, 8 waves/block, one WMMA/wave/step.
// Each lane owns exactly ONE (batch, hidden) state pair -> 5-exp gate chain.
// Branch-free, EXEC-manipulation-free steady state (t=0 peeled), single
// LDS-only barrier per step, hA ping-pong, x prefetched (scalarized) ahead.
// WMMA rows 8..15 are don't-care: lanes lhi==1 dump them into z-slice rows
// 8..15, which the gate readback (rows 0..7) never touches.
// ---------------------------------------------------------------------------
__global__ void __launch_bounds__(256, 1) ar_lstm(
    const float* __restrict__ xT, const float* __restrict__ u,
    const float* __restrict__ v, const float* __restrict__ w0,
    const _Float16* __restrict__ WhB, _Float16* __restrict__ hs) {
  __shared__ float zs[8 * 256];                    // per-wave 16-row z slices
  __shared__ alignas(32) _Float16 hA[2][512];      // h, f16 A-operand layout

  const int tid  = threadIdx.x;
  const int w    = tid >> 5;        // wave id 0..7 -> hidden units [4w,4w+4)
  const int lane = tid & 31;
  const int ln   = lane & 15;
  const int lhi  = lane >> 4;
  const int bbase = blockIdx.x * 8; // 8 batches per block

  // persistent B operand + rank-1 coefficients (per permuted gate column)
  const v16h Bw = *(const v16h*)(WhB + w * 512 + lane * 16);
  const int  kg = (ln & 3) * Dn + 4 * w + (ln >> 2);
  const float u_l = u[kg], v_l = v[kg], w0_l = w0[kg];

  // gate ownership: lane -> (batch gb = lane>>2, hidden k0 = 4w + (lane&3))
  const int gb = lane >> 2;
  const int k0 = 4 * w + (lane & 3);
  const int wlane2 = gb + 16 * ((k0 >> 3) & 1);      // hA write position
  const int whalf  = (k0 & 7) + 8 * (k0 >> 4);
  float* const zsw = zs + w * 256;
  const float* const zq = zsw + gb * 16 + (lane & 3) * 4;  // i,f,g,o quad
  float creg = 0.f;

  // x register prefetch: one 32B row slice (batches bbase..bbase+7)
  const float* const xrow = xT + bbase;
  v8f xc = *(const v8f*)xrow;                      // x[:,0], used at t=1

  // ---- t = 0: h == 0 so z = w0 (no WMMA, no hA init needed) ----
  {
#pragma unroll
    for (int vv = 0; vv < 8; ++vv)
      zsw[(vv + 8 * lhi) * 16 + ln] = w0_l;        // rows 8..15 = don't-care
    const float zi = zq[0], zf = zq[1], zg = zq[2], zo = zq[3];
    creg = fast_sigmoid(zf) * creg + fast_sigmoid(zi) * fast_tanh(zg);
    const _Float16 hh = (_Float16)(fast_sigmoid(zo) * fast_tanh(creg));
    hA[1][wlane2 * 16 + whalf] = hh;
    lds_fence_signal();
    hs[((size_t)bbase + gb) * Dn + k0] = hh;
    bar_wait();
  }

  // ---- steady state: branch-free, full-EXEC throughout ----
#pragma unroll 2
  for (int t = 1; t < Sn; ++t) {
    const int rb = t & 1;
    const v8f xn = *(const v8f*)(xrow + (size_t)t * Bn);  // feeds step t+1
    const v16h a = *(const v16h*)(hA[rb] + lane * 16);
    v8f cz = {};
    cz = __builtin_amdgcn_wmma_f32_16x16x32_f16(false, a, false, Bw,
                                                (short)0, cz, false, false);
#pragma unroll
    for (int vv = 0; vv < 8; ++vv)
      zsw[(vv + 8 * lhi) * 16 + ln] = cz[vv] + fmaf(xc[vv], u_l, v_l);
    const float zi = zq[0], zf = zq[1], zg = zq[2], zo = zq[3];
    creg = fast_sigmoid(zf) * creg + fast_sigmoid(zi) * fast_tanh(zg);
    const _Float16 hh = (_Float16)(fast_sigmoid(zo) * fast_tanh(creg));
    hA[rb ^ 1][wlane2 * 16 + whalf] = hh;            // next step's A operand
    lds_fence_signal();                              // LDS visible, arrive
    hs[((size_t)t * Bn + bbase + gb) * Dn + k0] = hh;  // off critical path
    xc = xn;                                         // commit prefetch
    bar_wait();
  }
}

// ---------------------------------------------------------------------------
// Output projection: logits[b,s,c] = hs[s,b,:] @ W_out + b_out.
// Memory-bound: 256 MB f32 out (streaming, non-temporal); hs (16 MB) from L2.
// ---------------------------------------------------------------------------
__global__ void __launch_bounds__(256) ar_out(
    const _Float16* __restrict__ hs, const _Float16* __restrict__ WoutB,
    const float* __restrict__ b_out, float* __restrict__ out) {
  const int tid  = threadIdx.x;
  const int w    = tid >> 5;
  const int lane = tid & 31;
  const int ln   = lane & 15;
  const int lhi  = lane >> 4;

  const int gw    = blockIdx.x * 8 + w;   // 0..16383
  const int s     = gw >> 1;
  const int bbase = (gw & 1) * 16;

  // gather A operand (16 rows x K=32 f16) in ISA layout: two 16B runs per lane
  const _Float16* ap =
      hs + ((size_t)s * Bn + bbase + ln) * Dn + (size_t)lhi * 8;
  const v8h lo = *(const v8h*)ap;         // K = 8*lhi .. +7
  const v8h hi = *(const v8h*)(ap + 16);  // K = 16 + 8*lhi .. +7
  v16h a;
#pragma unroll
  for (int j = 0; j < 8; ++j) { a[j] = lo[j]; a[8 + j] = hi[j]; }

#pragma unroll 4
  for (int nt = 0; nt < 16; ++nt) {
    const v16h bb = *(const v16h*)(WoutB + nt * 512 + lane * 16);
    v8f c = {};
    c = __builtin_amdgcn_wmma_f32_16x16x32_f16(false, a, false, bb,
                                               (short)0, c, false, false);
    const float bo = b_out[nt * 16 + ln];
#pragma unroll
    for (int vv = 0; vv < 8; ++vv) {
      const int b = bbase + vv + 8 * lhi;
      const size_t idx = (size_t)b * Sn * Cn + (size_t)s * Cn + nt * 16 + ln;
      __builtin_nontemporal_store(c[vv] + bo, out + idx);
    }
  }
}

// ---------------------------------------------------------------------------
extern "C" void kernel_launch(void* const* d_in, const int* in_sizes, int n_in,
                              void* d_out, int out_size, void* d_ws,
                              size_t ws_size, hipStream_t stream) {
  (void)in_sizes; (void)n_in; (void)out_size; (void)ws_size;
  const float* x      = (const float*)d_in[0];
  const float* bos    = (const float*)d_in[1];
  const float* W_in   = (const float*)d_in[2];
  const float* b_in   = (const float*)d_in[3];
  const float* Wx     = (const float*)d_in[4];
  const float* Wh     = (const float*)d_in[5];
  const float* b_lstm = (const float*)d_in[6];
  const float* W_out  = (const float*)d_in[7];
  const float* b_out  = (const float*)d_in[8];

  char* ws = (char*)d_ws;
  float*     u     = (float*)(ws + 0);
  float*     v     = (float*)(ws + 512);
  float*     w0    = (float*)(ws + 1024);
  _Float16*  WhB   = (_Float16*)(ws + 2048);            // 8 KB
  _Float16*  WoutB = (_Float16*)(ws + 2048 + 8192);     // 16 KB
  float*     xT    = (float*)(ws + 32768);              // 1 MB (S*B f32)
  _Float16*  hs    = (_Float16*)(ws + 32768 + 1048576); // 16 MB (S*B*D f16)
  float*     out   = (float*)d_out;

  ar_prep<<<1, 256, 0, stream>>>(bos, W_in, b_in, Wx, Wh, b_lstm, W_out,
                                 u, v, w0, WhB, WoutB);
  ar_xt<<<(Bn * Sn) / 256, 256, 0, stream>>>(x, xT);
  ar_lstm<<<4, 256, 0, stream>>>(xT, u, v, w0, WhB, hs);
  ar_out<<<2048, 256, 0, stream>>>(hs, WoutB, b_out, out);
}
